// SelfAttention_55173149885065
// MI455X (gfx1250) — compile-verified, matrix operand order checked by reading, not run on hardware
//
#include <hip/hip_runtime.h>

// ---------------------------------------------------------------------------
// Problem constants (match reference)
// ---------------------------------------------------------------------------
constexpr int Bc   = 2;
constexpr int Sc   = 2048;
constexpr int HIDc = 2048;
constexpr int Hc   = 16;
constexpr int KVc  = 4;
constexpr int Dc   = 128;
constexpr int ROTc = 64;
constexpr int Mrows = Bc * Sc;          // 4096 token rows
constexpr int NQG   = 2 * HIDc;         // 4096 (q + gate)
constexpr int NKV   = KVc * Dc;         // 512

// ---------------------------------------------------------------------------
// Types for WMMA
// ---------------------------------------------------------------------------
typedef __attribute__((ext_vector_type(8)))  __bf16 bf16x8;
typedef __attribute__((ext_vector_type(16))) __bf16 bf16x16;
typedef __attribute__((ext_vector_type(8)))  float  v8f;

// ---------------------------------------------------------------------------
// Workspace layout (bytes)
// ---------------------------------------------------------------------------
constexpr size_t SZ_HSB = (size_t)Mrows * HIDc * 2;        // hs bf16 (M,K)
constexpr size_t SZ_WQT = (size_t)NQG * HIDc * 2;          // Wq^T bf16 (N,K)
constexpr size_t SZ_WKT = (size_t)NKV * HIDc * 2;
constexpr size_t SZ_WVT = (size_t)NKV * HIDc * 2;
constexpr size_t SZ_WOT = (size_t)HIDc * HIDc * 2;
constexpr size_t SZ_QG  = (size_t)Mrows * NQG * 4;         // qg fp32
constexpr size_t SZ_KP  = (size_t)Mrows * NKV * 4;         // k proj fp32
constexpr size_t SZ_VP  = (size_t)Mrows * NKV * 4;
constexpr size_t SZ_QB  = (size_t)Bc * Hc * Sc * Dc * 2;   // roped Q bf16 (B,H,S,D)
constexpr size_t SZ_KB  = (size_t)Bc * KVc * Sc * Dc * 2;  // roped K bf16 (B,KV,S,D)
constexpr size_t SZ_VT  = (size_t)Bc * KVc * Dc * Sc * 2;  // V^T bf16 (B,KV,D,S)
constexpr size_t SZ_O   = (size_t)Mrows * HIDc * 4;        // attn out fp32 (B,S,H*D)
constexpr size_t SZ_O2  = (size_t)Mrows * HIDc * 2;        // normed/gated bf16

constexpr size_t OFF_HSB = 0;
constexpr size_t OFF_WQT = OFF_HSB + SZ_HSB;
constexpr size_t OFF_WKT = OFF_WQT + SZ_WQT;
constexpr size_t OFF_WVT = OFF_WKT + SZ_WKT;
constexpr size_t OFF_WOT = OFF_WVT + SZ_WVT;
constexpr size_t OFF_QG  = OFF_WOT + SZ_WOT;
constexpr size_t OFF_KP  = OFF_QG  + SZ_QG;
constexpr size_t OFF_VP  = OFF_KP  + SZ_KP;
constexpr size_t OFF_QB  = OFF_VP  + SZ_VP;
constexpr size_t OFF_KB  = OFF_QB  + SZ_QB;
constexpr size_t OFF_VT  = OFF_KB  + SZ_KB;
constexpr size_t OFF_O   = OFF_VT  + SZ_VT;
constexpr size_t OFF_O2  = OFF_O   + SZ_O;

// ---------------------------------------------------------------------------
// Fragment loader: 16-bit A/B matrix fragment per the CDNA5 ISA striping.
// Row (A: M, B: N) = lane & 15. K elements: [k0+koff .. +7] and [k0+koff+16 ..+7]
// where koff = 8 for lanes 16-31. Two 16-byte loads per fragment.
// Works for both global and LDS pointers (addrspace inferred).
// ---------------------------------------------------------------------------
__device__ __forceinline__ bf16x16 load_frag(const __bf16* base,
                                             int stride, int lane, int k0) {
  int m    = lane & 15;
  int koff = (lane & 16) >> 1;   // 0 or 8
  const __bf16* p = base + (size_t)m * stride + k0 + koff;
  bf16x8 lo = *reinterpret_cast<const bf16x8*>(p);
  bf16x8 hi = *reinterpret_cast<const bf16x8*>(p + 16);
  bf16x16 f;
#pragma unroll
  for (int i = 0; i < 8; ++i) { f[i] = lo[i]; f[i + 8] = hi[i]; }
  return f;
}

__device__ __forceinline__ v8f wmma_bf16(bf16x16 a, bf16x16 b, v8f c) {
  return __builtin_amdgcn_wmma_f32_16x16x32_bf16(false, a, false, b,
                                                 (short)0, c, false, false);
}

// Async global->LDS 16B copy (per-lane addresses), GVS addressing:
//   mem = SGPR64_base + VGPR32_offset ; LDS dest = VGPR32 lds byte address.
// Tracked by ASYNCcnt.
__device__ __forceinline__ void async_ld16(unsigned ldsoff,
                                           unsigned long long sbase,
                                           unsigned voff) {
  asm volatile("global_load_async_to_lds_b128 %0, %1, %2"
               :: "v"(ldsoff), "v"(voff), "s"(sbase) : "memory");
}

// ---------------------------------------------------------------------------
// fp32 -> bf16 elementwise convert
// ---------------------------------------------------------------------------
__global__ void f32_to_bf16_kernel(const float* __restrict__ in,
                                   unsigned short* __restrict__ out_, size_t n) {
  __bf16* out = reinterpret_cast<__bf16*>(out_);
  size_t i = (size_t)blockIdx.x * blockDim.x + threadIdx.x;
  if (i < n) out[i] = (__bf16)in[i];
}

// fp32 (K,N) -> bf16 (N,K) transpose convert
__global__ void transpose_f32_to_bf16_kernel(const float* __restrict__ in,
                                             unsigned short* __restrict__ out_,
                                             int Kdim, int Ndim) {
  __bf16* out = reinterpret_cast<__bf16*>(out_);
  size_t i = (size_t)blockIdx.x * blockDim.x + threadIdx.x;
  if (i >= (size_t)Kdim * Ndim) return;
  int n = (int)(i % Ndim);
  int k = (int)(i / Ndim);
  out[(size_t)n * Kdim + k] = (__bf16)in[i];
}

// ---------------------------------------------------------------------------
// WMMA GEMM v2: C(M,N) fp32 = A(M,K) bf16 x Bt(N,K) bf16.
// Block: 256 threads = 8 waves; block tile 128M x 128N; wave tile 32M x 64N.
// A/B K-slabs (128x32 bf16 each) staged in LDS with double-buffered
// GLOBAL_LOAD_ASYNC_TO_LDS_B128 (ASYNCcnt), overlapped with WMMA.
// Requires M%128==0, N%128==0, K%32==0.
// ---------------------------------------------------------------------------
__global__ __launch_bounds__(256) void gemm_bf16_v2(
    const unsigned short* __restrict__ A_, const unsigned short* __restrict__ Bt_,
    float* __restrict__ C, int Mdim, int Ndim, int Kdim) {
  const __bf16* A  = reinterpret_cast<const __bf16*>(A_);
  const __bf16* Bt = reinterpret_cast<const __bf16*>(Bt_);

  __shared__ __align__(16) __bf16 ldsA[2][128 * 32];
  __shared__ __align__(16) __bf16 ldsB[2][128 * 32];

  int t    = threadIdx.x;
  int lane = t & 31;
  int wv   = t >> 5;              // 0..7
  int mw   = wv >> 1;             // 0..3  (32-row slice)
  int nw   = wv & 1;              // 0..1  (64-col slice)
  int nblk = Ndim >> 7;
  int mbase = (int)(blockIdx.x / nblk) * 128;
  int nbase = (int)(blockIdx.x % nblk) * 128;

  int srow = t >> 2;              // 0..63 staging row
  int scc  = (t & 3) << 3;        // staging col chunk (elements)
  const int nk = Kdim >> 5;

  auto issue_tile = [&](int k, int buf) {
    int kcol = k << 5;
#pragma unroll
    for (int rr = 0; rr < 2; ++rr) {
      int row = srow + rr * 64;
      unsigned lo = (unsigned)(uintptr_t)&ldsA[buf][row * 32 + scc];
      unsigned go = (unsigned)(((size_t)(mbase + row) * Kdim + kcol + scc) * 2);
      async_ld16(lo, (unsigned long long)(uintptr_t)A, go);
    }
#pragma unroll
    for (int rr = 0; rr < 2; ++rr) {
      int row = srow + rr * 64;
      unsigned lo = (unsigned)(uintptr_t)&ldsB[buf][row * 32 + scc];
      unsigned go = (unsigned)(((size_t)(nbase + row) * Kdim + kcol + scc) * 2);
      async_ld16(lo, (unsigned long long)(uintptr_t)Bt, go);
    }
  };

  v8f acc[2][4] = {};

  issue_tile(0, 0);
  for (int k = 0; k < nk; ++k) {
    int buf = k & 1;
    if (k + 1 < nk) {
      issue_tile(k + 1, buf ^ 1);
      asm volatile("s_wait_asynccnt 4" ::: "memory");   // tile k landed
    } else {
      asm volatile("s_wait_asynccnt 0" ::: "memory");
    }
    __syncthreads();                                    // all waves' slab ready

    bf16x16 af0 = load_frag(&ldsA[buf][(mw * 32)      * 32], 32, lane, 0);
    bf16x16 af1 = load_frag(&ldsA[buf][(mw * 32 + 16) * 32], 32, lane, 0);
#pragma unroll
    for (int j = 0; j < 4; ++j) {
      bf16x16 bfr = load_frag(&ldsB[buf][(nw * 64 + j * 16) * 32], 32, lane, 0);
      acc[0][j] = wmma_bf16(af0, bfr, acc[0][j]);
      acc[1][j] = wmma_bf16(af1, bfr, acc[1][j]);
    }
    __syncthreads();                                    // done reading this buf
  }

  int mrowoff = (lane >> 4) << 3;
  int ncol    = lane & 15;
#pragma unroll
  for (int i = 0; i < 2; ++i)
#pragma unroll
    for (int j = 0; j < 4; ++j)
#pragma unroll
      for (int r = 0; r < 8; ++r)
        C[(size_t)(mbase + mw * 32 + i * 16 + mrowoff + r) * Ndim +
          nbase + nw * 64 + j * 16 + ncol] = acc[i][j][r];
}

// ---------------------------------------------------------------------------
// RoPE pack for Q: qg fp32 (B*S, 4096) -> qb bf16 (B,H,S,D)
// ---------------------------------------------------------------------------
__global__ void pack_q_kernel(const float* __restrict__ qg,
                              const float* __restrict__ cosb,
                              const float* __restrict__ sinb,
                              unsigned short* __restrict__ qb_) {
  __bf16* qb = reinterpret_cast<__bf16*>(qb_);
  size_t idx = (size_t)blockIdx.x * blockDim.x + threadIdx.x;
  if (idx >= (size_t)Bc * Sc * Hc * Dc) return;
  int d = (int)(idx % Dc); size_t t = idx / Dc;
  int h = (int)(t % Hc);   t /= Hc;
  int s = (int)(t % Sc);
  int b = (int)(t / Sc);
  const float* row = qg + ((size_t)b * Sc + s) * NQG + (size_t)h * Dc;
  float val;
  if (d < ROTc) {
    float cs = cosb[((size_t)b * Sc + s) * ROTc + d];
    float sn = sinb[((size_t)b * Sc + s) * ROTc + d];
    float qpass = row[ROTc + d];
    float rh = (d < ROTc / 2) ? -row[d + ROTc / 2] : row[d - ROTc / 2];
    val = cs * qpass + rh * sn;
  } else {
    val = row[d];
  }
  qb[(((size_t)b * Hc + h) * Sc + s) * Dc + d] = (__bf16)val;
}

// RoPE pack for K (sin used for both terms, per reference) + V transpose.
__global__ void pack_kv_kernel(const float* __restrict__ kp,
                               const float* __restrict__ vp,
                               const float* __restrict__ sinb,
                               unsigned short* __restrict__ kb_,
                               unsigned short* __restrict__ vt_) {
  __bf16* kb = reinterpret_cast<__bf16*>(kb_);
  __bf16* vt = reinterpret_cast<__bf16*>(vt_);
  size_t idx = (size_t)blockIdx.x * blockDim.x + threadIdx.x;
  if (idx >= (size_t)Bc * Sc * KVc * Dc) return;
  int d = (int)(idx % Dc); size_t t = idx / Dc;
  int g = (int)(t % KVc);  t /= KVc;
  int s = (int)(t % Sc);
  int b = (int)(t / Sc);
  const float* krow = kp + ((size_t)b * Sc + s) * NKV + (size_t)g * Dc;
  float kval;
  if (d < ROTc) {
    float sn = sinb[((size_t)b * Sc + s) * ROTc + d];
    float kpass = krow[ROTc + d];
    float rh = (d < ROTc / 2) ? -krow[d + ROTc / 2] : krow[d - ROTc / 2];
    kval = sn * kpass + rh * sn;
  } else {
    kval = krow[d];
  }
  kb[(((size_t)b * KVc + g) * Sc + s) * Dc + d] = (__bf16)kval;
  float vv = vp[((size_t)b * Sc + s) * NKV + (size_t)g * Dc + d];
  vt[(((size_t)b * KVc + g) * Dc + d) * Sc + s] = (__bf16)vv;
}

// ---------------------------------------------------------------------------
// Flash attention v2 (GQA-shared K/V):
// Block = 4 waves = the 4 query heads of one KV group, same 16-query tile.
// K (32x128) and V^T (128x32) slabs staged once per 32-key step into LDS with
// double-buffered async loads, shared by all 4 waves (4x global traffic cut).
// Per-wave: online softmax, LDS C->A relayout of P, WMMA QK^T and PV.
// ---------------------------------------------------------------------------
__global__ __launch_bounds__(128) void flash_attn_kernel(
    const unsigned short* __restrict__ qb_, const unsigned short* __restrict__ kb_,
    const unsigned short* __restrict__ vt_, float* __restrict__ obuf) {
  const __bf16* qb = reinterpret_cast<const __bf16*>(qb_);
  const __bf16* kb = reinterpret_cast<const __bf16*>(kb_);
  const __bf16* vt = reinterpret_cast<const __bf16*>(vt_);

  __shared__ __align__(16) __bf16 kt_lds[2][32 * 128];   // keys x d
  __shared__ __align__(16) __bf16 vt_lds[2][128 * 32];   // d x keys
  __shared__ __align__(16) __bf16 lds_p[4][16 * 32];     // per-wave P relayout

  int t    = threadIdx.x;
  int lane = t & 31;
  int wv   = t >> 5;                 // 0..3 : query head within KV group
  int qt = blockIdx.x;               // 0 .. S/16-1
  int bg = blockIdx.y;               // 0 .. B*KV-1
  int b = bg / KVc, g = bg % KVc;
  int h = g * (Hc / KVc) + wv;
  int q0 = qt * 16;

  const __bf16* Qbase  = qb + ((size_t)((size_t)b * Hc + h) * Sc + q0) * Dc;
  const __bf16* Kbase  = kb + ((size_t)((size_t)b * KVc + g) * Sc) * Dc;
  const __bf16* Vtbase = vt + ((size_t)((size_t)b * KVc + g) * Dc) * Sc;

  bf16x16 qfrag[4];
#pragma unroll
  for (int f = 0; f < 4; ++f) qfrag[f] = load_frag(Qbase, Dc, lane, f * 32);

  v8f oacc[8] = {};
  float mrow[8], lrow[8];
#pragma unroll
  for (int r = 0; r < 8; ++r) { mrow[r] = -1e30f; lrow[r] = 0.f; }

  const float scale = 0.088388347648318447f; // 1/sqrt(128)
  int half8 = (lane >> 4) << 3;  // 0 or 8
  int ncol  = lane & 15;
  int kend  = q0 + 16;                       // causal: keys 0..q0+15
  const int nsteps = (kend + 31) >> 5;

  // stage 32 keys of K (row-major, stride 128) and V^T (stride 32) into LDS
  auto issue_kv = [&](int step, int buf) {
    int k0 = step << 5;
#pragma unroll
    for (int i = 0; i < 4; ++i) {
      int chunk = t + i * 128;               // 0..511
      int row = chunk >> 4;                  // key 0..31
      int cc  = (chunk & 15) << 3;           // d chunk
      unsigned lo = (unsigned)(uintptr_t)&kt_lds[buf][row * 128 + cc];
      unsigned go = (unsigned)(((size_t)(k0 + row) * Dc + cc) * 2);
      async_ld16(lo, (unsigned long long)(uintptr_t)Kbase, go);
    }
#pragma unroll
    for (int i = 0; i < 4; ++i) {
      int chunk = t + i * 128;
      int row = chunk >> 2;                  // d 0..127
      int cc  = (chunk & 3) << 3;            // key chunk
      unsigned lo = (unsigned)(uintptr_t)&vt_lds[buf][row * 32 + cc];
      unsigned go = (unsigned)(((size_t)row * Sc + k0 + cc) * 2);
      async_ld16(lo, (unsigned long long)(uintptr_t)Vtbase, go);
    }
  };

  issue_kv(0, 0);
  for (int s = 0; s < nsteps; ++s) {
    int buf = s & 1;
    int k0 = s << 5;
    if (s + 1 < nsteps) {
      issue_kv(s + 1, buf ^ 1);
      asm volatile("s_wait_asynccnt 8" ::: "memory");   // slab s landed
    } else {
      asm volatile("s_wait_asynccnt 0" ::: "memory");
    }
    __syncthreads();                                    // slab visible to block

    float sc[2][8];
#pragma unroll
    for (int tt = 0; tt < 2; ++tt) {
      v8f acc = {};
#pragma unroll
      for (int f = 0; f < 4; ++f) {
        bf16x16 kf = load_frag(&kt_lds[buf][(16 * tt) * 128], 128, lane, f * 32);
        acc = wmma_bf16(qfrag[f], kf, acc);
      }
      int key = k0 + 16 * tt + ncol;
#pragma unroll
      for (int r = 0; r < 8; ++r) {
        int m = q0 + half8 + r;
        sc[tt][r] = acc[r] * scale + ((key <= m) ? 0.f : -1e9f);
      }
    }
#pragma unroll
    for (int r = 0; r < 8; ++r) {
      float vmax = fmaxf(sc[0][r], sc[1][r]);
#pragma unroll
      for (int off = 8; off >= 1; off >>= 1)
        vmax = fmaxf(vmax, __shfl_xor(vmax, off, 32));
      float mnew  = fmaxf(mrow[r], vmax);
      float alpha = __expf(mrow[r] - mnew);
      float p0 = __expf(sc[0][r] - mnew);
      float p1 = __expf(sc[1][r] - mnew);
      sc[0][r] = p0; sc[1][r] = p1;
      float psum = p0 + p1;
#pragma unroll
      for (int off = 8; off >= 1; off >>= 1)
        psum += __shfl_xor(psum, off, 32);
      lrow[r] = lrow[r] * alpha + psum;
      mrow[r] = mnew;
#pragma unroll
      for (int j = 0; j < 8; ++j) oacc[j][r] *= alpha;
    }
    // P: C-layout -> A-layout via per-wave LDS tile
#pragma unroll
    for (int tt = 0; tt < 2; ++tt)
#pragma unroll
      for (int r = 0; r < 8; ++r)
        lds_p[wv][(half8 + r) * 32 + 16 * tt + ncol] = (__bf16)sc[tt][r];
    __syncthreads();
    bf16x16 pfrag;
    {
      int m    = lane & 15;
      int koff = (lane & 16) >> 1;
      const __bf16* p = &lds_p[wv][m * 32 + koff];
      bf16x8 lo = *reinterpret_cast<const bf16x8*>(p);
      bf16x8 hi = *reinterpret_cast<const bf16x8*>(p + 16);
#pragma unroll
      for (int i = 0; i < 8; ++i) { pfrag[i] = lo[i]; pfrag[i + 8] = hi[i]; }
    }
#pragma unroll
    for (int j = 0; j < 8; ++j) {
      bf16x16 vf = load_frag(&vt_lds[buf][(16 * j) * 32], 32, lane, 0);
      oacc[j] = wmma_bf16(pfrag, vf, oacc[j]);
    }
    __syncthreads();                                    // block done with buf
  }

#pragma unroll
  for (int r = 0; r < 8; ++r) {
    float inv = 1.0f / lrow[r];
    int m = q0 + half8 + r;
    float* orow = obuf + ((size_t)b * Sc + m) * (size_t)(Hc * Dc) + (size_t)h * Dc;
#pragma unroll
    for (int j = 0; j < 8; ++j)
      orow[16 * j + ncol] = oacc[j][r] * inv;
  }
}

// ---------------------------------------------------------------------------
// RMSNorm * (1+w) * sigmoid(gate) -> bf16
// ---------------------------------------------------------------------------
__global__ __launch_bounds__(256) void rmsnorm_gate_kernel(
    const float* __restrict__ obuf, const float* __restrict__ qg,
    const float* __restrict__ nw, unsigned short* __restrict__ o2_) {
  __bf16* o2 = reinterpret_cast<__bf16*>(o2_);
  int row = blockIdx.x;   // 0 .. B*S-1
  const float* x    = obuf + (size_t)row * HIDc;
  const float* gate = qg   + (size_t)row * NQG + HIDc;
  __shared__ float red[256];
  float ss = 0.f;
  for (int c = threadIdx.x; c < HIDc; c += 256) { float v = x[c]; ss += v * v; }
  red[threadIdx.x] = ss;
  __syncthreads();
  for (int s = 128; s > 0; s >>= 1) {
    if ((int)threadIdx.x < s) red[threadIdx.x] += red[threadIdx.x + s];
    __syncthreads();
  }
  float rinv = rsqrtf(red[0] / (float)HIDc + 1e-6f);
  for (int c = threadIdx.x; c < HIDc; c += 256) {
    float sg = 1.f / (1.f + __expf(-gate[c]));
    o2[(size_t)row * HIDc + c] = (__bf16)(x[c] * rinv * (1.f + nw[c]) * sg);
  }
}

// ---------------------------------------------------------------------------
// Launch
// ---------------------------------------------------------------------------
extern "C" void kernel_launch(void* const* d_in, const int* in_sizes, int n_in,
                              void* d_out, int out_size, void* d_ws, size_t ws_size,
                              hipStream_t stream) {
  const float* hs   = (const float*)d_in[0];
  const float* cosb = (const float*)d_in[1];
  const float* sinb = (const float*)d_in[2];
  // d_in[3] attention_mask: causal -1e9 mask reproduced analytically
  const float* Wq = (const float*)d_in[4];
  const float* Wk = (const float*)d_in[5];
  const float* Wv = (const float*)d_in[6];
  const float* Wo = (const float*)d_in[7];
  const float* nw = (const float*)d_in[8];
  float* out = (float*)d_out;

  char* ws = (char*)d_ws;
  unsigned short* hsb = (unsigned short*)(ws + OFF_HSB);
  unsigned short* wqt = (unsigned short*)(ws + OFF_WQT);
  unsigned short* wkt = (unsigned short*)(ws + OFF_WKT);
  unsigned short* wvt = (unsigned short*)(ws + OFF_WVT);
  unsigned short* wot = (unsigned short*)(ws + OFF_WOT);
  float*          qg  = (float*)(ws + OFF_QG);
  float*          kp  = (float*)(ws + OFF_KP);
  float*          vp  = (float*)(ws + OFF_VP);
  unsigned short* qb  = (unsigned short*)(ws + OFF_QB);
  unsigned short* kbb = (unsigned short*)(ws + OFF_KB);
  unsigned short* vtb = (unsigned short*)(ws + OFF_VT);
  float*          ob  = (float*)(ws + OFF_O);
  unsigned short* o2  = (unsigned short*)(ws + OFF_O2);

  // 1) precision conversion / weight transposes
  {
    size_t n = (size_t)Mrows * HIDc;
    f32_to_bf16_kernel<<<(unsigned)((n + 255) / 256), 256, 0, stream>>>(hs, hsb, n);
  }
  transpose_f32_to_bf16_kernel<<<(unsigned)(((size_t)HIDc * NQG  + 255) / 256), 256, 0, stream>>>(Wq, wqt, HIDc, NQG);
  transpose_f32_to_bf16_kernel<<<(unsigned)(((size_t)HIDc * NKV  + 255) / 256), 256, 0, stream>>>(Wk, wkt, HIDc, NKV);
  transpose_f32_to_bf16_kernel<<<(unsigned)(((size_t)HIDc * NKV  + 255) / 256), 256, 0, stream>>>(Wv, wvt, HIDc, NKV);
  transpose_f32_to_bf16_kernel<<<(unsigned)(((size_t)HIDc * HIDc + 255) / 256), 256, 0, stream>>>(Wo, wot, HIDc, HIDc);

  // 2) projection GEMMs (WMMA + async-LDS double buffering)
  gemm_bf16_v2<<<(Mrows / 128) * (NQG / 128), 256, 0, stream>>>(hsb, wqt, qg, Mrows, NQG, HIDc);
  gemm_bf16_v2<<<(Mrows / 128) * (NKV / 128), 256, 0, stream>>>(hsb, wkt, kp, Mrows, NKV, HIDc);
  gemm_bf16_v2<<<(Mrows / 128) * (NKV / 128), 256, 0, stream>>>(hsb, wvt, vp, Mrows, NKV, HIDc);

  // 3) RoPE + layout packing
  {
    size_t n = (size_t)Bc * Sc * Hc * Dc;
    pack_q_kernel<<<(unsigned)((n + 255) / 256), 256, 0, stream>>>(qg, cosb, sinb, qb);
  }
  {
    size_t n = (size_t)Bc * Sc * KVc * Dc;
    pack_kv_kernel<<<(unsigned)((n + 255) / 256), 256, 0, stream>>>(kp, vp, sinb, kbb, vtb);
  }

  // 4) flash attention: 4 waves share one KV group's K/V slabs
  flash_attn_kernel<<<dim3(Sc / 16, Bc * KVc), 128, 0, stream>>>(qb, kbb, vtb, ob);

  // 5) RMSNorm * sigmoid(gate)
  rmsnorm_gate_kernel<<<Mrows, 256, 0, stream>>>(ob, qg, nw, o2);

  // 6) output projection -> d_out (fp32)
  gemm_bf16_v2<<<(Mrows / 128) * (HIDc / 128), 256, 0, stream>>>(o2, wot, out, Mrows, HIDc, HIDc);
}
